// MultiheadDilatedAttention_6725918785579
// MI455X (gfx1250) — compile-verified, matrix operand order checked by reading, not run on hardware
//
#include <hip/hip_runtime.h>

// ---------------------------------------------------------------------------
// MI455X (gfx1250) LongNet dilated attention, bf16 WMMA pipeline + TDM staging.
// BATCH=2, SEQ=8192, EMBED=1024, HEADS=16, HEAD_DIM=64
// groups: (r,s) in {(1,1024),(2,2048),(4,4096),(8,8192)} -> dilated len 1024 each
// ---------------------------------------------------------------------------

typedef unsigned short u16;
typedef __attribute__((ext_vector_type(16))) __bf16     v16bf;
typedef __attribute__((ext_vector_type(8)))  float      v8f;
typedef __attribute__((ext_vector_type(4)))  unsigned   u32x4;
typedef __attribute__((ext_vector_type(4)))  float      f32x4;
typedef __attribute__((ext_vector_type(4)))  int        i32x4;
typedef __attribute__((ext_vector_type(8)))  int        i32x8;

#define NTOK   16384   // BATCH * SEQ
#define EMBED  1024

#if __has_builtin(__builtin_amdgcn_tensor_load_to_lds)
#define USE_TDM 1
#else
#define USE_TDM 0
#endif

__device__ __forceinline__ u16 f2bf(float f) {
  union { float f; unsigned u; } v; v.f = f;
  unsigned r = v.u + 0x7FFFu + ((v.u >> 16) & 1u);   // round-to-nearest-even
  return (u16)(r >> 16);
}
__device__ __forceinline__ float bf2f(u16 h) {
  union { unsigned u; float f; } v; v.u = ((unsigned)h) << 16; return v.f;
}

union Frag16 { u32x4 q[2]; v16bf v; };   // 16 bf16 = one WMMA A/B operand

__device__ __forceinline__ v8f wmma_bf16(const Frag16& a, const Frag16& b, v8f c) {
  return __builtin_amdgcn_wmma_f32_16x16x32_bf16(
      false, a.v, false, b.v, (short)0, c, false, false);
}
__device__ __forceinline__ v8f v8f_zero() {
  v8f z = {0.f, 0.f, 0.f, 0.f, 0.f, 0.f, 0.f, 0.f};
  return z;
}

// ---------------------------------------------------------------------------
// TDM: DMA a 128(N) x 32(K) bf16 tile of W[1024,1024] into LDS, with hardware
// row padding (64B data + 16B pad per row == u16[40] row stride).
// ---------------------------------------------------------------------------
#if USE_TDM
__device__ __forceinline__ void tdm_load_w_tile(const u16* W, int n0, int k0,
                                                void* ldsDst) {
  unsigned long long ga =
      (unsigned long long)W + (((unsigned long long)n0 << 10) + (unsigned)k0) * 2ull;
  // LDS aperture base has zero low 32 bits -> low 32 bits of generic VA == LDS offset
  unsigned lds = (unsigned)(unsigned long long)ldsDst;

  u32x4 g0;
  g0[0] = 1u;                                        // count=1, user descriptor
  g0[1] = lds;                                       // lds_addr
  g0[2] = (unsigned)(ga & 0xFFFFFFFFull);            // global_addr[31:0]
  g0[3] = (unsigned)((ga >> 32) & 0x01FFFFFFull) | 0x80000000u;  // addr[56:32]|type=2

  i32x8 g1;
  g1[0] = (int)((1u << 16)        // data_size = 2 bytes
              | (1u << 20)        // pad_enable
              | (3u << 22)        // pad_interval: every 16 DWORDs (=64B row)
              | (3u << 25));      // pad_amount: 4 DWORDs (=16B)
  g1[1] = (int)(1024u << 16);     // tensor_dim0[15:0]=1024 in bits[63:48]
  g1[2] = (int)(1024u << 16);     // dim0[31:16]=0, tensor_dim1[15:0]=1024
  g1[3] = (int)(32u << 16);       // dim1[31:16]=0, tile_dim0=32 (K)
  g1[4] = 128;                    // tile_dim1=128 (N rows), tile_dim2=0
  g1[5] = 1024;                   // tensor_dim0_stride = 1024 elems
  g1[6] = 0;
  g1[7] = 0;

  i32x4 z4 = {0, 0, 0, 0};
#if defined(__clang_major__) && (__clang_major__ >= 23)
  i32x8 z8 = {0, 0, 0, 0, 0, 0, 0, 0};
  __builtin_amdgcn_tensor_load_to_lds(g0, g1, z4, z4, z8, 0);
#else
  __builtin_amdgcn_tensor_load_to_lds(g0, g1, z4, z4, 0);
#endif
}
#endif

// ---------------------------------------------------------------------------
// fp32 -> bf16 elementwise (weights)
// ---------------------------------------------------------------------------
__global__ void cvt_kernel(const float* __restrict__ src, u16* __restrict__ dst, int n) {
  int i = blockIdx.x * blockDim.x + threadIdx.x;
  if (i < n) dst[i] = f2bf(src[i]);
}

__global__ void zero_kernel(u32x4* __restrict__ dst, int n4) {
  int i = blockIdx.x * blockDim.x + threadIdx.x;
  u32x4 z = {0u, 0u, 0u, 0u};
  if (i < n4) dst[i] = z;
}

// ---------------------------------------------------------------------------
// GEMM: C[M,1024] = A[M,1024] @ W[1024,1024]^T + bias
//   A_IS_F32 : A is fp32 (converted to bf16 while staging into LDS)
//   OUT_F32  : store fp32 (final projection) vs bf16 (QKV intermediates)
// Block tile 128x128, K-step 32; 8 waves as 4(M)x2(N); each wave 32x64 out.
// W tile staged by the Tensor Data Mover (double-buffered, wave 0 issues).
// ---------------------------------------------------------------------------
template <bool A_IS_F32, bool OUT_F32>
__global__ __launch_bounds__(256) void gemm_wmma(
    const void* __restrict__ Aptr, const u16* __restrict__ W,
    const float* __restrict__ bias, void* __restrict__ Cptr) {
  __shared__ u16 sA[128][40];        // +8 halves pad, keeps 16B alignment
  __shared__ u16 sB2[2][128][40];    // double-buffered W tile (TDM destination)

  const int tid   = threadIdx.x;
  const int lane  = tid & 31;
  const int wv    = tid >> 5;        // 0..7
  const int waveM = wv >> 1;         // 0..3
  const int waveN = wv & 1;          // 0..1
  const int m0 = blockIdx.y * 128;
  const int n0 = blockIdx.x * 128;
  const int lr = lane & 15;          // row / col within 16-tile
  const int hf = lane >> 4;          // half-wave select (K sub-pattern)

  v8f acc[2][4];
#pragma unroll
  for (int i = 0; i < 2; ++i)
#pragma unroll
    for (int j = 0; j < 4; ++j) acc[i][j] = v8f_zero();

#if USE_TDM
  if (wv == 0) tdm_load_w_tile(W, n0, 0, &sB2[0][0][0]);
#endif

  const int NK = 32;                 // 1024 / 32 K-steps
  for (int i = 0; i < NK; ++i) {
    const int k0 = i * 32;
    // ---- stage A (128x32) into LDS as bf16
    if (A_IS_F32) {
      const float* A = (const float*)Aptr;
#pragma unroll
      for (int it = 0; it < 4; ++it) {
        int idx = tid + it * 256;                 // 1024 float4 chunks
        int row = idx >> 3;
        int c4  = idx & 7;
        f32x4 f = *(const f32x4*)(A + (size_t)(m0 + row) * 1024 + k0 + c4 * 4);
        u16* d = &sA[row][c4 * 4];
        d[0] = f2bf(f.x); d[1] = f2bf(f.y); d[2] = f2bf(f.z); d[3] = f2bf(f.w);
      }
    } else {
      const u16* A = (const u16*)Aptr;
#pragma unroll
      for (int it = 0; it < 2; ++it) {
        int idx = tid + it * 256;                 // 512 16B chunks
        int row = idx >> 2;
        int c8  = idx & 3;
        u32x4 d = *(const u32x4*)(A + (size_t)(m0 + row) * 1024 + k0 + c8 * 8);
        *(u32x4*)&sA[row][c8 * 8] = d;
      }
    }

#if USE_TDM
    // ---- TDM: kick next W tile, ensure current tile landed
    if (wv == 0) {
      if (i + 1 < NK) {
        tdm_load_w_tile(W, n0, (i + 1) * 32, &sB2[(i + 1) & 1][0][0]);
        __builtin_amdgcn_s_wait_tensorcnt((short)1);   // tile i complete
      } else {
        __builtin_amdgcn_s_wait_tensorcnt((short)0);
      }
    }
    const u16(*sB)[40] = sB2[i & 1];
#else
    // ---- manual W tile staging fallback
#pragma unroll
    for (int it = 0; it < 2; ++it) {
      int idx = tid + it * 256;
      int row = idx >> 2;
      int c8  = idx & 3;
      u32x4 d = *(const u32x4*)(W + (size_t)(n0 + row) * 1024 + k0 + c8 * 8);
      *(u32x4*)&sB2[0][row][c8 * 8] = d;
    }
    const u16(*sB)[40] = sB2[0];
#endif
    __syncthreads();

    // ---- fragments + 8 WMMAs
    Frag16 a[2], b[4];
#pragma unroll
    for (int ms = 0; ms < 2; ++ms) {
      int r = waveM * 32 + ms * 16 + lr;
      a[ms].q[0] = *(const u32x4*)&sA[r][8 * hf];
      a[ms].q[1] = *(const u32x4*)&sA[r][16 + 8 * hf];
    }
#pragma unroll
    for (int ns = 0; ns < 4; ++ns) {
      int r = waveN * 64 + ns * 16 + lr;
      b[ns].q[0] = *(const u32x4*)&sB[r][8 * hf];
      b[ns].q[1] = *(const u32x4*)&sB[r][16 + 8 * hf];
    }
#pragma unroll
    for (int ms = 0; ms < 2; ++ms)
#pragma unroll
      for (int ns = 0; ns < 4; ++ns)
        acc[ms][ns] = wmma_bf16(a[ms], b[ns], acc[ms][ns]);
    __syncthreads();
  }

  // ---- epilogue: C layout = lane holds col (lane&15), rows g + 8*(lane>>4)
#pragma unroll
  for (int ms = 0; ms < 2; ++ms) {
#pragma unroll
    for (int ns = 0; ns < 4; ++ns) {
      int col = n0 + waveN * 64 + ns * 16 + lr;
      float bv = bias[col];
#pragma unroll
      for (int g = 0; g < 8; ++g) {
        int row = m0 + waveM * 32 + ms * 16 + g + 8 * hf;
        float v = acc[ms][ns][g] + bv;
        if (OUT_F32) ((float*)Cptr)[(size_t)row * 1024 + col] = v;
        else         ((u16*)Cptr)[(size_t)row * 1024 + col]   = f2bf(v);
      }
    }
  }
}

// ---------------------------------------------------------------------------
// Dilated flash attention: one wave per (group, batch, seg, head, 16-q tile).
// All groups have dilated window = 1024 tokens, head_dim = 64.
// Wave-tile prefix: g0=4096, g1=2048, g2=1024, g3=512  (total 7680).
// ---------------------------------------------------------------------------
__global__ __launch_bounds__(128) void dilated_attn_wmma(
    const u16* __restrict__ Q, const u16* __restrict__ K,
    const u16* __restrict__ V, u16* __restrict__ O) {
  __shared__ u16 sP[4][16][40];    // per-wave P tile: 16 q-rows x 32 keys (+pad)
  __shared__ u16 sVT[4][64][40];   // per-wave V^T tile: 64 d-rows x 32 keys (+pad)

  const int lane = threadIdx.x & 31;
  const int wv   = threadIdx.x >> 5;
  const int wid  = blockIdx.x * 4 + wv;
  const int lr   = lane & 15;
  const int hf   = lane >> 4;

  int g;
  if (wid < 4096) g = 0; else if (wid < 6144) g = 1; else if (wid < 7168) g = 2; else g = 3;
  const int pref4[4] = {0, 4096, 6144, 7168};
  int rem = wid - pref4[g];
  const int qtile = rem & 63;  rem >>= 6;
  const int hin   = rem & 3;   rem >>= 2;
  const int nslog = 3 - g;                       // log2(nseg)
  const int seg   = rem & ((1 << nslog) - 1);
  const int b     = rem >> nslog;
  const int slen  = 1024 << g;                   // segment length
  const int r     = 1 << g;                      // dilation
  const int off   = g & (r - 1);                 // i % r
  const int head  = g * 4 + hin;
  const size_t bbase = (size_t)b * 8192;

  auto rowptr = [&](const u16* base, int j) -> const u16* {
    int t = seg * slen + off + j * r;            // dilated token index
    return base + ((bbase + t) * 1024 + head * 64);
  };

  // Q fragments: 16 rows x 64 d -> two A-frags (K=0..31, 32..63)
  Frag16 qa[2];
  {
    const u32x4* qr = (const u32x4*)rowptr(Q, qtile * 16 + lr);
    qa[0].q[0] = qr[hf];     qa[0].q[1] = qr[2 + hf];
    qa[1].q[0] = qr[4 + hf]; qa[1].q[1] = qr[6 + hf];
  }

  v8f o[4];
#pragma unroll
  for (int i = 0; i < 4; ++i) o[i] = v8f_zero();
  float mrow[8], lsum[8];
#pragma unroll
  for (int i = 0; i < 8; ++i) { mrow[i] = -1e30f; lsum[i] = 0.f; }

  for (int kt = 0; kt < 32; ++kt) {
    // ---- prefetch next key tile (one dilated row per lane, 128B each)
    if (kt + 1 < 32) {
      __builtin_prefetch(rowptr(K, (kt + 1) * 32 + lane), 0, 0);
      __builtin_prefetch(rowptr(V, (kt + 1) * 32 + lane), 0, 0);
    }

    // ---- scores S[16q x 32k] as two C-frags (2 WMMAs each over d=64)
    v8f sc[2];
#pragma unroll
    for (int kb = 0; kb < 2; ++kb) {
      Frag16 kf0, kf1;
      const u32x4* kr = (const u32x4*)rowptr(K, kt * 32 + kb * 16 + lr);
      kf0.q[0] = kr[hf];     kf0.q[1] = kr[2 + hf];
      kf1.q[0] = kr[4 + hf]; kf1.q[1] = kr[6 + hf];
      v8f c = v8f_zero();
      c = wmma_bf16(qa[0], kf0, c);
      c = wmma_bf16(qa[1], kf1, c);
      sc[kb] = c;
    }

    // ---- stage V^T (lane loads its key row, scatters d into LDS columns)
    {
      const u32x4* vr = (const u32x4*)rowptr(V, kt * 32 + lane);
      u32x4 vbuf[8];
#pragma unroll
      for (int i = 0; i < 8; ++i) vbuf[i] = vr[i];
      const u16* vs = (const u16*)vbuf;
#pragma unroll
      for (int d = 0; d < 64; ++d) sVT[wv][d][lane] = vs[d];
    }

    // ---- online softmax (C layout: lane = key col lr, rows g + 8*hf)
#pragma unroll
    for (int gi = 0; gi < 8; ++gi) {
      float s0 = sc[0][gi] * 0.125f;             // 1/sqrt(64)
      float s1 = sc[1][gi] * 0.125f;
      float m = fmaxf(s0, s1);
      m = fmaxf(m, __shfl_xor(m, 1, 32));
      m = fmaxf(m, __shfl_xor(m, 2, 32));
      m = fmaxf(m, __shfl_xor(m, 4, 32));
      m = fmaxf(m, __shfl_xor(m, 8, 32));
      float mnew = fmaxf(mrow[gi], m);
      float corr = __expf(mrow[gi] - mnew);
      mrow[gi] = mnew;
      float p0 = __expf(s0 - mnew);
      float p1 = __expf(s1 - mnew);
      float ps = p0 + p1;
      ps += __shfl_xor(ps, 1, 32);
      ps += __shfl_xor(ps, 2, 32);
      ps += __shfl_xor(ps, 4, 32);
      ps += __shfl_xor(ps, 8, 32);
      lsum[gi] = lsum[gi] * corr + ps;
#pragma unroll
      for (int dt = 0; dt < 4; ++dt) o[dt][gi] *= corr;
      sP[wv][gi + 8 * hf][lr]      = f2bf(p0);   // key block 0
      sP[wv][gi + 8 * hf][16 + lr] = f2bf(p1);   // key block 1
    }

    // ---- O += P x V  (A-frag from sP, B-frags from sVT; LDS pipe is in-order)
    Frag16 pa;
    pa.q[0] = *(const u32x4*)&sP[wv][lr][8 * hf];
    pa.q[1] = *(const u32x4*)&sP[wv][lr][16 + 8 * hf];
#pragma unroll
    for (int dt = 0; dt < 4; ++dt) {
      Frag16 vb;
      vb.q[0] = *(const u32x4*)&sVT[wv][dt * 16 + lr][8 * hf];
      vb.q[1] = *(const u32x4*)&sVT[wv][dt * 16 + lr][16 + 8 * hf];
      o[dt] = wmma_bf16(pa, vb, o[dt]);
    }
  }

  // ---- epilogue: scale 1/l * 1/num_groups, scatter into dilated layout
#pragma unroll
  for (int gi = 0; gi < 8; ++gi) {
    float inv = 0.25f / lsum[gi];
    int j = qtile * 16 + gi + 8 * hf;
    int t = seg * slen + off + j * r;
    u16* orow = O + ((bbase + t) * 1024 + head * 64);
#pragma unroll
    for (int dt = 0; dt < 4; ++dt)
      orow[dt * 16 + lr] = f2bf(o[dt][gi] * inv);
  }
}

// ---------------------------------------------------------------------------
// LayerNorm over 1024 features, one block per token row
// ---------------------------------------------------------------------------
__global__ __launch_bounds__(256) void layernorm_kernel(
    const u16* __restrict__ X, const float* __restrict__ gamma,
    const float* __restrict__ beta, u16* __restrict__ Y) {
  __shared__ float redS[8], redQ[8];
  const int row = blockIdx.x;
  const int tid = threadIdx.x;
  const u16* x = X + (size_t)row * 1024;
  float v[4], s = 0.f, ss = 0.f;
#pragma unroll
  for (int i = 0; i < 4; ++i) {
    float f = bf2f(x[tid * 4 + i]);
    v[i] = f; s += f; ss += f * f;
  }
#pragma unroll
  for (int o = 1; o < 32; o <<= 1) {
    s  += __shfl_xor(s,  o, 32);
    ss += __shfl_xor(ss, o, 32);
  }
  if ((tid & 31) == 0) { redS[tid >> 5] = s; redQ[tid >> 5] = ss; }
  __syncthreads();
  float ts = 0.f, tq = 0.f;
#pragma unroll
  for (int i = 0; i < 8; ++i) { ts += redS[i]; tq += redQ[i]; }
  float mu   = ts * (1.f / 1024.f);
  float var  = tq * (1.f / 1024.f) - mu * mu;
  float rstd = rsqrtf(var + 1e-5f);
#pragma unroll
  for (int i = 0; i < 4; ++i) {
    int c = tid * 4 + i;
    Y[(size_t)row * 1024 + c] = f2bf((v[i] - mu) * rstd * gamma[c] + beta[c]);
  }
}

// ---------------------------------------------------------------------------
// Host side
// ---------------------------------------------------------------------------
extern "C" void kernel_launch(void* const* d_in, const int* in_sizes, int n_in,
                              void* d_out, int out_size, void* d_ws, size_t ws_size,
                              hipStream_t stream) {
  (void)in_sizes; (void)n_in; (void)out_size; (void)ws_size;
  const float* query = (const float*)d_in[0];
  const float* key_  = (const float*)d_in[1];
  const float* value = (const float*)d_in[2];
  const float* Wq = (const float*)d_in[3];  const float* bq = (const float*)d_in[4];
  const float* Wk = (const float*)d_in[5];  const float* bk = (const float*)d_in[6];
  const float* Wv = (const float*)d_in[7];  const float* bv = (const float*)d_in[8];
  const float* gamma = (const float*)d_in[9];
  const float* beta  = (const float*)d_in[10];
  const float* Wo = (const float*)d_in[11]; const float* bo = (const float*)d_in[12];
  float* out = (float*)d_out;

  // workspace carve-up (bf16 elements): 4 weights + 5 activation buffers ~176MB
  const size_t WELEM = (size_t)1024 * 1024;
  const size_t XELEM = (size_t)NTOK * 1024;
  u16* wqb  = (u16*)d_ws;
  u16* wkb  = wqb + WELEM;
  u16* wvb  = wkb + WELEM;
  u16* wob  = wvb + WELEM;
  u16* qb   = wob + WELEM;
  u16* kb   = qb + XELEM;
  u16* vb   = kb + XELEM;
  u16* attn = vb + XELEM;
  u16* lnb  = attn + XELEM;

  // 1) weights -> bf16
  cvt_kernel<<<(int)(WELEM / 256), 256, 0, stream>>>(Wq, wqb, (int)WELEM);
  cvt_kernel<<<(int)(WELEM / 256), 256, 0, stream>>>(Wk, wkb, (int)WELEM);
  cvt_kernel<<<(int)(WELEM / 256), 256, 0, stream>>>(Wv, wvb, (int)WELEM);
  cvt_kernel<<<(int)(WELEM / 256), 256, 0, stream>>>(Wo, wob, (int)WELEM);
  // attention output must start zeroed (dilation leaves holes per head group)
  zero_kernel<<<(int)(XELEM / 8 / 256), 256, 0, stream>>>((u32x4*)attn, (int)(XELEM / 8));

  // 2) QKV projections (fp32 activations, bf16 out)
  dim3 ggrid(8, 128);   // N/128 x M/128
  gemm_wmma<true, false><<<ggrid, 256, 0, stream>>>(query, wqb, bq, qb);
  gemm_wmma<true, false><<<ggrid, 256, 0, stream>>>(key_,  wkb, bk, kb);
  gemm_wmma<true, false><<<ggrid, 256, 0, stream>>>(value, wvb, bv, vb);

  // 3) dilated flash attention: 7680 wave-tiles, 4 waves/block
  dilated_attn_wmma<<<1920, 128, 0, stream>>>(qb, kb, vb, attn);

  // 4) LayerNorm
  layernorm_kernel<<<NTOK, 256, 0, stream>>>(attn, gamma, beta, lnb);

  // 5) output projection (bf16 in, fp32 out)
  gemm_wmma<false, true><<<ggrid, 256, 0, stream>>>(lnb, wob, bo, out);
}
